// JointNet_15625091023613
// MI455X (gfx1250) — compile-verified, matrix operand order checked by reading, not run on hardware
//
#include <hip/hip_runtime.h>
#include <hip/hip_fp16.h>

typedef _Float16 v16h __attribute__((ext_vector_type(16)));
typedef float    v8f  __attribute__((ext_vector_type(8)));

union V16HU { v16h v; uint4 q[2]; };

// ---------------- prep: Y[R,1024] = X[R,512] @ W[512,1024] + b ----------------
__global__ __launch_bounds__(256) void proj_kernel(
    const float* __restrict__ X, const float* __restrict__ W,
    const float* __restrict__ bias, float* __restrict__ Y)
{
  __shared__ float xt[16][64];
  const int tid = threadIdx.x;
  const int r0  = blockIdx.x * 16;
  const int f   = blockIdx.y * 256 + tid;
  float acc[16];
#pragma unroll
  for (int m = 0; m < 16; ++m) acc[m] = 0.f;

  for (int dc = 0; dc < 512; dc += 64) {
    __syncthreads();
#pragma unroll
    for (int e = 0; e < 4; ++e) {
      int li = tid + e * 256;
      int m = li >> 6, d = li & 63;
      xt[m][d] = X[(size_t)(r0 + m) * 512 + dc + d];
    }
    __syncthreads();
#pragma unroll 8
    for (int d = 0; d < 64; ++d) {
      float w = W[(size_t)(dc + d) * 1024 + f];
#pragma unroll
      for (int m = 0; m < 16; ++m) acc[m] = fmaf(xt[m][d], w, acc[m]);
    }
  }
  const float bb = bias[f];
#pragma unroll
  for (int m = 0; m < 16; ++m) Y[(size_t)(r0 + m) * 1024 + f] = acc[m] + bb;
}

// -------- W_fc [K=1024][V=1024] f32 -> WT [V][K] f16 (N-major for WMMA B) --------
__global__ __launch_bounds__(256) void twfc_kernel(
    const float* __restrict__ W, _Float16* __restrict__ WT)
{
  __shared__ float tile[32][33];
  const int tx = threadIdx.x & 31;
  const int ty = threadIdx.x >> 5;      // 0..7
  const int k0 = blockIdx.x * 32;
  const int v0 = blockIdx.y * 32;
#pragma unroll
  for (int r = 0; r < 4; ++r) {
    int k = ty + r * 8;
    tile[k][tx] = W[(size_t)(k0 + k) * 1024 + v0 + tx];
  }
  __syncthreads();
#pragma unroll
  for (int r = 0; r < 4; ++r) {
    int v = ty + r * 8;
    WT[(size_t)(v0 + v) * 1024 + k0 + tx] = (_Float16)tile[tx][v];
  }
}

// ---------------- fused joint: tanh(pe+pd) @ WfcT + b, log_softmax, mask ----------------
// Block: 512 threads = 16 waves. Each wave owns a DISTINCT 64-column N-slice and
// computes BOTH 16-row M-tiles against every B fragment it loads (B read once/block).
union SMem {
  _Float16 xs[32][1024];  // 64 KB A-stage (f16 tanh(pe+pd))
  float    red[16][32];   // cross-wave softmax reduction (overlaid after GEMM)
};

__global__ __launch_bounds__(512) void joint_kernel(
    const float* __restrict__ pe, const float* __restrict__ pd,
    const _Float16* __restrict__ WT, const float* __restrict__ bfc,
    const int* __restrict__ encL, const int* __restrict__ decL,
    float* __restrict__ out)
{
  __shared__ SMem sm;
  const int tid  = threadIdx.x;
  const int lane = tid & 31;
  const int wave = tid >> 5;     // 0..15 : N slice of 64 columns

  const int r0 = blockIdx.x * 32;      // 32 joint cells per block (same b,t)
  const int u0 = r0 & 63;
  const int t  = (r0 >> 6) & 255;
  const int b  = r0 >> 14;

  const float* peR = pe + (size_t)(b * 256 + t) * 1024;
  const float* pdB = pd + (size_t)(b * 64 + u0) * 1024;

  // ---- stage A = tanh(pe + pd) as f16 into LDS: 32 rows x 1024 ----
#pragma unroll 4
  for (int e = 0; e < 64; ++e) {
    int idx = tid + e * 512;
    int m = idx >> 10, k = idx & 1023;
    sm.xs[m][k] = (_Float16)tanhf(peR[k] + pdB[(size_t)m * 1024 + k]);
  }
  __syncthreads();

  const int mrow  = lane & 15;
  const int khalf = (lane >> 4) << 3;   // 0 or 8 (ISA 16-bit A/B K layout)
  const int n0    = wave << 6;          // 64-col slice

  const _Float16* Arow0 = &sm.xs[mrow][0];
  const _Float16* Arow1 = &sm.xs[16 + mrow][0];
  const _Float16* Bbase = WT + (size_t)(n0 + mrow) * 1024;

  v8f acc[2][4];
  const v8f vzero = {0.f, 0.f, 0.f, 0.f, 0.f, 0.f, 0.f, 0.f};
#pragma unroll
  for (int h = 0; h < 2; ++h)
#pragma unroll
    for (int j = 0; j < 4; ++j) acc[h][j] = vzero;

#pragma unroll 2
  for (int k0 = 0; k0 < 1024; k0 += 32) {
    V16HU A0, A1;
    {
      const _Float16* ap0 = Arow0 + k0 + khalf;
      A0.q[0] = *(const uint4*)(ap0);
      A0.q[1] = *(const uint4*)(ap0 + 16);
      const _Float16* ap1 = Arow1 + k0 + khalf;
      A1.q[0] = *(const uint4*)(ap1);
      A1.q[1] = *(const uint4*)(ap1 + 16);
    }
#pragma unroll
    for (int j = 0; j < 4; ++j) {
      V16HU Bm;
      const _Float16* bp = Bbase + (size_t)j * 16 * 1024 + k0 + khalf;
      Bm.q[0] = *(const uint4*)(bp);
      Bm.q[1] = *(const uint4*)(bp + 16);
      acc[0][j] = __builtin_amdgcn_wmma_f32_16x16x32_f16(
          false, A0.v, false, Bm.v, (short)0, acc[0][j], false, false);
      acc[1][j] = __builtin_amdgcn_wmma_f32_16x16x32_f16(
          false, A1.v, false, Bm.v, (short)0, acc[1][j], false, false);
    }
  }

  // ---- + b_fc ----
#pragma unroll
  for (int j = 0; j < 4; ++j) {
    float bv = bfc[n0 + j * 16 + mrow];
#pragma unroll
    for (int h = 0; h < 2; ++h)
#pragma unroll
      for (int i = 0; i < 8; ++i) acc[h][j][i] += bv;
  }

  const int rbase = (lane >> 4) << 3;   // C/D layout: lanes 16-31 hold rows 8..15

  // ---- row max: over 4 tiles, then 16 lanes, then 16 N-waves via LDS ----
  float pm[2][8];
#pragma unroll
  for (int h = 0; h < 2; ++h)
#pragma unroll
    for (int i = 0; i < 8; ++i) {
      float v = acc[h][0][i];
#pragma unroll
      for (int j = 1; j < 4; ++j) v = fmaxf(v, acc[h][j][i]);
      pm[h][i] = v;
    }
#pragma unroll
  for (int s = 1; s <= 8; s <<= 1)
#pragma unroll
    for (int h = 0; h < 2; ++h)
#pragma unroll
      for (int i = 0; i < 8; ++i)
        pm[h][i] = fmaxf(pm[h][i], __shfl_xor(pm[h][i], s, 32));
  __syncthreads();              // all xs reads done; safe to overlay sm.red
  if ((lane & 15) == 0) {
#pragma unroll
    for (int h = 0; h < 2; ++h)
#pragma unroll
      for (int i = 0; i < 8; ++i) sm.red[wave][h * 16 + rbase + i] = pm[h][i];
  }
  __syncthreads();
  float rowmax[2][8];
#pragma unroll
  for (int h = 0; h < 2; ++h)
#pragma unroll
    for (int i = 0; i < 8; ++i) {
      float v = sm.red[0][h * 16 + rbase + i];
#pragma unroll
      for (int w = 1; w < 16; ++w) v = fmaxf(v, sm.red[w][h * 16 + rbase + i]);
      rowmax[h][i] = v;
    }
  __syncthreads();

  // ---- sum of exp ----
  float ps[2][8];
#pragma unroll
  for (int h = 0; h < 2; ++h)
#pragma unroll
    for (int i = 0; i < 8; ++i) {
      float v = 0.f;
#pragma unroll
      for (int j = 0; j < 4; ++j) v += __expf(acc[h][j][i] - rowmax[h][i]);
      ps[h][i] = v;
    }
#pragma unroll
  for (int s = 1; s <= 8; s <<= 1)
#pragma unroll
    for (int h = 0; h < 2; ++h)
#pragma unroll
      for (int i = 0; i < 8; ++i) ps[h][i] += __shfl_xor(ps[h][i], s, 32);
  if ((lane & 15) == 0) {
#pragma unroll
    for (int h = 0; h < 2; ++h)
#pragma unroll
      for (int i = 0; i < 8; ++i) sm.red[wave][h * 16 + rbase + i] = ps[h][i];
  }
  __syncthreads();
  float logZ[2][8];
#pragma unroll
  for (int h = 0; h < 2; ++h)
#pragma unroll
    for (int i = 0; i < 8; ++i) {
      float v = sm.red[0][h * 16 + rbase + i];
#pragma unroll
      for (int w = 1; w < 16; ++w) v += sm.red[w][h * 16 + rbase + i];
      logZ[h][i] = rowmax[h][i] + __logf(v);
    }

  // ---- masked store: out[r, n] = ok ? logit - logZ : 0 ----
  const int eL = encL[b], dL = decL[b];
  const bool tok = (t < eL);
  float* outR = out + (size_t)r0 * 1024;
#pragma unroll
  for (int h = 0; h < 2; ++h)
#pragma unroll
    for (int i = 0; i < 8; ++i) {
      int m = h * 16 + rbase + i;
      bool ok = tok && ((u0 + m) < dL);
      float lz = logZ[h][i];
#pragma unroll
      for (int j = 0; j < 4; ++j) {
        float v = ok ? (acc[h][j][i] - lz) : 0.f;
        outR[(size_t)m * 1024 + n0 + j * 16 + mrow] = v;
      }
    }
}

extern "C" void kernel_launch(void* const* d_in, const int* in_sizes, int n_in,
                              void* d_out, int out_size, void* d_ws, size_t ws_size,
                              hipStream_t stream) {
  (void)in_sizes; (void)n_in; (void)out_size; (void)ws_size;
  const float* enc  = (const float*)d_in[0];
  const float* dec  = (const float*)d_in[1];
  const float* Wenc = (const float*)d_in[2];
  const float* benc = (const float*)d_in[3];
  const float* Wdec = (const float*)d_in[4];
  const float* bdec = (const float*)d_in[5];
  const float* Wfc  = (const float*)d_in[6];
  const float* bfc  = (const float*)d_in[7];
  const int*   encL = (const int*)d_in[8];
  const int*   decL = (const int*)d_in[9];
  float* out = (float*)d_out;

  char* ws = (char*)d_ws;
  float*    pe = (float*)(ws);                         // [1024][1024] f32 = 4 MB
  float*    pd = (float*)(ws + (size_t)(4u << 20));    // [ 256][1024] f32 = 1 MB
  _Float16* WT = (_Float16*)(ws + (size_t)(5u << 20)); // [1024][1024] f16 = 2 MB

  dim3 blk256(256);
  proj_kernel<<<dim3(1024 / 16, 4), blk256, 0, stream>>>(enc, Wenc, benc, pe);
  proj_kernel<<<dim3(256 / 16, 4),  blk256, 0, stream>>>(dec, Wdec, bdec, pd);
  twfc_kernel<<<dim3(32, 32), blk256, 0, stream>>>(Wfc, WT);
  joint_kernel<<<dim3(65536 / 32), dim3(512), 0, stream>>>(
      pe, pd, WT, bfc, encL, decL, out);
}
